// SAE_5798205849983
// MI455X (gfx1250) — compile-verified, matrix operand order checked by reading, not run on hardware
//
#include <hip/hip_runtime.h>

// SAE forward for MI455X (gfx1250): bf16 WMMA encoder + exact top-k + fp32 sparse decode.
#define SAE_B 8192
#define SAE_D 256
#define SAE_H 8192
#define SAE_TOPK 32
#define BLOCK_M 64          // rows per block
#define XSTRIDE 264         // x tile LDS row stride (bf16 elems), 256 + 8 pad
#define TSTRIDE 129         // pre_acts tile LDS row stride (f32 elems), 128 + 1 pad

typedef __bf16 bf16_t;
typedef __attribute__((ext_vector_type(16))) __bf16          bf16x16;
typedef __attribute__((ext_vector_type(8)))  float           f32x8;
typedef __attribute__((ext_vector_type(4)))  float           f32x4;
typedef __attribute__((ext_vector_type(4)))  unsigned int    u32x4;
typedef __attribute__((ext_vector_type(4)))  unsigned short  u16x4;

static __device__ __forceinline__ unsigned short f2bf_bits(float f) {
    unsigned u = __builtin_bit_cast(unsigned, f);
    u += 0x7FFFu + ((u >> 16) & 1u);          // round-to-nearest-even
    return (unsigned short)(u >> 16);
}

union BfPack { struct { u32x4 lo, hi; } u; bf16x16 v; };

// Load a 16-element bf16 fragment: 8 bf16 at p, 8 bf16 at p+16 (ISA 7.12.2 layout).
static __device__ __forceinline__ bf16x16 load_frag(const bf16_t* p) {
    BfPack pk;
    pk.u.lo = *(const u32x4*)(p);
    pk.u.hi = *(const u32x4*)(p + 16);
    return pk.v;
}

// ---------------- W_enc fp32 -> bf16 pre-conversion -------------------------
__global__ __launch_bounds__(256)
void sae_wenc_to_bf16(const float* __restrict__ w, bf16_t* __restrict__ o, int n4) {
    int i = blockIdx.x * 256 + threadIdx.x;
    if (i < n4) {
        f32x4 v = *(const f32x4*)(w + (size_t)i * 4);
        u16x4 r;
        r.x = f2bf_bits(v.x);
        r.y = f2bf_bits(v.y);
        r.z = f2bf_bits(v.z);
        r.w = f2bf_bits(v.w);
        *(u16x4*)(o + (size_t)i * 4) = r;
    }
}

// ---------------- Fused encoder GEMM + top-k + sparse decode ----------------
__global__ __launch_bounds__(256)
void sae_fused_kernel(const float* __restrict__ x,
                      const bf16_t* __restrict__ wenc_bf,
                      const float* __restrict__ b_enc,
                      const float* __restrict__ W_dec,
                      const float* __restrict__ b_dec,
                      float* __restrict__ out) {
    __shared__ __align__(16) bf16_t lds_x[BLOCK_M * XSTRIDE];      // 33792 B
    __shared__ __align__(16) float  lds_tile[BLOCK_M * TSTRIDE];   // 33024 B
    __shared__ float cand_val[BLOCK_M * 64];                       // 16384 B (2 parts x 32)
    __shared__ int   cand_idx[BLOCK_M * 64];                       // 16384 B

    const int tid  = threadIdx.x;
    const int lane = tid & 31;
    const int wv   = tid >> 5;                 // wave 0..7
    const int R0   = blockIdx.x * BLOCK_M;

    // init candidate lists (val=0 is a valid neutral: relu output >= 0, and a
    // (0, idx) entry contributes nothing to the decode)
    for (int i = tid; i < BLOCK_M * 64; i += 256) { cand_val[i] = 0.0f; cand_idx[i] = 0; }

    // load x tile, subtract b_dec, convert to bf16 into padded LDS
    for (int e = tid; e < BLOCK_M * SAE_D / 4; e += 256) {
        int r  = e >> 6;             // 64 chunks of 4 per row
        int c4 = (e & 63) << 2;
        f32x4 xv = *(const f32x4*)(x + (size_t)(R0 + r) * SAE_D + c4);
        f32x4 bd = *(const f32x4*)(b_dec + c4);
        u16x4 pk;
        pk.x = f2bf_bits(xv.x - bd.x);
        pk.y = f2bf_bits(xv.y - bd.y);
        pk.z = f2bf_bits(xv.z - bd.z);
        pk.w = f2bf_bits(xv.w - bd.w);
        *(u16x4*)(&lds_x[r * XSTRIDE + c4]) = pk;
    }
    __syncthreads();

    // incremental top-32 merge threads: 2 per row, thread-private running min
    const bool merge_on = (tid < BLOCK_M * 2);
    const int  mrow  = tid >> 1;
    const int  mpart = tid & 1;
    float*     mylist = &cand_val[(mrow * 2 + mpart) * 32];
    int*       myidx  = &cand_idx[(mrow * 2 + mpart) * 32];
    float      curmin = 0.0f;
    int        minpos = 0;

    const int hoff = lane & 15;                 // N / M index within fragment
    const int koff = (lane < 16) ? 0 : 8;       // ISA 7.12.2 half-wave K split
    const int rbase = (lane >> 4) << 3;         // C/D: lanes 16-31 hold M+8

    for (int t = 0; t < SAE_H / 128; ++t) {
        const int h0 = t * 128 + wv * 16;
        f32x8 acc[4];
        #pragma unroll
        for (int rt = 0; rt < 4; ++rt) acc[rt] = (f32x8)0.0f;

        const bf16_t* bptr = wenc_bf + (size_t)(h0 + hoff) * SAE_D;
        #pragma unroll
        for (int kc = 0; kc < 8; ++kc) {
            const int d = kc * 32;
            bf16x16 bfrag = load_frag(bptr + d + koff);      // B: W_enc row h0+hoff
            #pragma unroll
            for (int rt = 0; rt < 4; ++rt) {
                bf16x16 afrag = load_frag(&lds_x[(rt * 16 + hoff) * XSTRIDE + d + koff]);
                acc[rt] = __builtin_amdgcn_wmma_f32_16x16x32_bf16(
                    false, afrag, false, bfrag, (short)0, acc[rt], false, false);
            }
        }

        // bias + relu, spill to LDS tile [64 rows][128 cols]
        const float bias = b_enc[h0 + hoff];
        const int   col  = wv * 16 + hoff;
        #pragma unroll
        for (int rt = 0; rt < 4; ++rt) {
            #pragma unroll
            for (int i = 0; i < 8; ++i) {
                float v = fmaxf(acc[rt][i] + bias, 0.0f);
                lds_tile[(rt * 16 + rbase + i) * TSTRIDE + col] = v;
            }
        }
        __syncthreads();

        if (merge_on) {
            const int cbase = mpart * 64;
            for (int j = 0; j < 64; ++j) {
                float v = lds_tile[mrow * TSTRIDE + cbase + j];
                if (v > curmin) {
                    mylist[minpos] = v;
                    myidx[minpos]  = t * 128 + cbase + j;
                    float mn = mylist[0]; int mp = 0;
                    for (int s = 1; s < 32; ++s) {
                        float q = mylist[s];
                        if (q < mn) { mn = q; mp = s; }
                    }
                    curmin = mn; minpos = mp;
                }
            }
        }
        __syncthreads();
    }

    // exact top-32 from the 64-candidate union; store into lds_tile row scratch
    if (tid < BLOCK_M) {
        const int row = tid;
        float* cv = &cand_val[row * 64];
        int*   ci = &cand_idx[row * 64];
        float* fv = &lds_tile[row * TSTRIDE];
        int*   fi = (int*)&lds_tile[row * TSTRIDE + 32];
        for (int s = 0; s < SAE_TOPK; ++s) {
            float best = -1.0f; int bp = 0;
            for (int j = 0; j < 64; ++j) {
                float q = cv[j];
                if (q > best) { best = q; bp = j; }
            }
            fv[s] = (best > 0.0f) ? best : 0.0f;
            fi[s] = ci[bp];
            cv[bp] = -1.0f;
        }
    }
    __syncthreads();

    // sparse decode: 16 threads per row, 16 f32 columns each, fp32 exact
    const int rloc  = tid >> 4;                 // 0..15
    const int dbase = (tid & 15) * 16;
    for (int rg = 0; rg < 4; ++rg) {
        const int row = rg * 16 + rloc;
        f32x4 a0 = *(const f32x4*)(b_dec + dbase + 0);
        f32x4 a1 = *(const f32x4*)(b_dec + dbase + 4);
        f32x4 a2 = *(const f32x4*)(b_dec + dbase + 8);
        f32x4 a3 = *(const f32x4*)(b_dec + dbase + 12);
        const float* fv = &lds_tile[row * TSTRIDE];
        const int*   fi = (const int*)&lds_tile[row * TSTRIDE + 32];
        for (int j = 0; j < SAE_TOPK; ++j) {
            const float v  = fv[j];
            const float* wd = W_dec + (size_t)fi[j] * SAE_D + dbase;
            a0 += v * (*(const f32x4*)(wd + 0));
            a1 += v * (*(const f32x4*)(wd + 4));
            a2 += v * (*(const f32x4*)(wd + 8));
            a3 += v * (*(const f32x4*)(wd + 12));
        }
        float* op = out + (size_t)(R0 + row) * SAE_D + dbase;
        *(f32x4*)(op + 0)  = a0;
        *(f32x4*)(op + 4)  = a1;
        *(f32x4*)(op + 8)  = a2;
        *(f32x4*)(op + 12) = a3;
    }
}

extern "C" void kernel_launch(void* const* d_in, const int* in_sizes, int n_in,
                              void* d_out, int out_size, void* d_ws, size_t ws_size,
                              hipStream_t stream) {
    const float* x     = (const float*)d_in[0];
    const float* W_enc = (const float*)d_in[1];
    const float* b_enc = (const float*)d_in[2];
    const float* W_dec = (const float*)d_in[3];
    const float* b_dec = (const float*)d_in[4];
    float* out = (float*)d_out;
    bf16_t* wenc_bf = (bf16_t*)d_ws;            // 8192*256*2 = 4 MB scratch

    const int n4 = SAE_H * SAE_D / 4;           // 524288 float4 chunks
    sae_wenc_to_bf16<<<(n4 + 255) / 256, 256, 0, stream>>>(W_enc, wenc_bf, n4);

    sae_fused_kernel<<<SAE_B / BLOCK_M, 256, 0, stream>>>(
        x, wenc_bf, b_enc, W_dec, b_dec, out);
}
